// Encoder_se3ACN_55911884259804
// MI455X (gfx1250) — compile-verified, matrix operand order checked by reading, not run on hardware
//
#include <hip/hip_runtime.h>
#include <hip/hip_bf16.h>
#include <math.h>

typedef __attribute__((ext_vector_type(16))) _Float16 v16h;
typedef __attribute__((ext_vector_type(8)))  _Float16 v8h;
typedef __attribute__((ext_vector_type(8)))  float    v8f;

#define NATOMS 286
#define NBATCH 2
#define HDIM   150
#define HP     160
#define MCHUNK 96
#define MAXR   3.0f

// softplus(5x)/5 using raw v_exp_f32 / v_log_f32 (base-2, no fixup code):
//   = max(x,0) + (ln2/5) * log2(1 + 2^(-|x| * 5*log2e))
// log2 argument is in [1,2] -> no denorm/inf paths needed.
__device__ __forceinline__ float sp5(float x) {
  float e2 = __builtin_amdgcn_exp2f(fabsf(x) * -7.213475204f); // exp(-|5x|)
  float lg = __builtin_amdgcn_logf(1.0f + e2);                 // log2(1+e)
  return fmaf(lg, 0.1386294361f, fmaxf(x, 0.0f));
}

// Load a 16x32 f16 fragment (A or B^T) from LDS, row-major with given stride.
// Per CDNA5 ISA: lanes 0-15 hold K {0..7, 16..23}, lanes 16-31 hold {8..15, 24..31}.
__device__ __forceinline__ v16h load_frag(const _Float16* base, int stride,
                                          int tile, int kt, int lane) {
  int row  = tile * 16 + (lane & 15);
  int koff = kt * 32 + ((lane & 16) ? 8 : 0);
  const _Float16* p = base + row * stride + koff;
  union { v16h v; struct { v8h lo; v8h hi; } s; } u;
  u.s.lo = *(const v8h*)(p);        // ds_load_b128
  u.s.hi = *(const v8h*)(p + 16);   // ds_load_b128
  return u.v;
}

template<bool SP, bool F16OUT>
__device__ __forceinline__ void store_tile(const v8f& acc, int mt, int nt,
                                           _Float16* o16, float* o32, int sO,
                                           int lane) {
  int row0 = mt * 16 + ((lane & 16) ? 8 : 0);
  int col  = nt * 16 + (lane & 15);
  #pragma unroll
  for (int v = 0; v < 8; ++v) {
    float x = acc[v];
    if (SP) x = sp5(x);
    if (F16OUT) o16[(row0 + v) * sO + col] = (_Float16)x;
    else        o32[(row0 + v) * sO + col] = x;
  }
}

// Tiled GEMM: M=96 (6 tiles) x K=KT*32 x N=TN*16 (TN even).
// 8 waves split output tiles; each wave does a 1x2 tile block sharing the A frag.
template<int KT, bool SP, bool F16OUT>
__device__ __forceinline__ void gemm_tiles(const _Float16* A, int sA,
                                           const _Float16* Bw, int sB,
                                           int TN,
                                           _Float16* o16, float* o32, int sO,
                                           int wave, int lane) {
  int TNh = TN >> 1;
  for (int t = wave; t < 6 * TNh; t += 8) {
    int mt = t / TNh, nt0 = (t % TNh) * 2, nt1 = nt0 + 1;
    v8f acc0 = {}, acc1 = {};
    #pragma unroll
    for (int kt = 0; kt < KT; ++kt) {
      v16h af = load_frag(A,  sA, mt,  kt, lane);
      v16h b0 = load_frag(Bw, sB, nt0, kt, lane);
      v16h b1 = load_frag(Bw, sB, nt1, kt, lane);
      acc0 = __builtin_amdgcn_wmma_f32_16x16x32_f16(false, af, false, b0,
                                                    (short)0, acc0, false, false);
      acc1 = __builtin_amdgcn_wmma_f32_16x16x32_f16(false, af, false, b1,
                                                    (short)0, acc1, false, false);
    }
    store_tile<SP, F16OUT>(acc0, mt, nt0, o16, o32, sO, lane);
    store_tile<SP, F16OUT>(acc1, mt, nt1, o16, o32, sO, lane);
  }
}

// f0 = emb[Z]
__global__ void gather_f0_kernel(const float* __restrict__ emb,
                                 const int* __restrict__ Z,
                                 float* __restrict__ f0, int n) {
  int i = blockIdx.x * blockDim.x + threadIdx.x;
  if (i < n) {
    int z = Z[i];
    #pragma unroll
    for (int k = 0; k < 4; ++k) f0[i * 4 + k] = emb[z * 4 + k];
  }
}

// One block per (b, a): radial-MLP chain on all neighbors j + fused masked einsum.
__global__ void __launch_bounds__(256)
conv_kernel(const float* __restrict__ xyz, const float* __restrict__ f_in,
            const float* __restrict__ w1, const float* __restrict__ w2,
            const float* __restrict__ w3, const float* __restrict__ wo,
            float* __restrict__ f_out, int din, int W /* = dout*din */) {
  extern __shared__ char smem[];
  _Float16* Wt2  = (_Float16*)smem;              // [160][160] transposed w2
  _Float16* Wt3  = Wt2 + HP * HP;                // [160][160] transposed w3
  _Float16* Wto  = Wt3 + HP * HP;                // [64][160]  transposed wo
  _Float16* Wt1  = Wto + 64 * HP;                // [160][32]  transposed w1
  _Float16* bufA = Wt1 + HP * 32;                // [96][160]  activations ping
  _Float16* bufB = bufA + MCHUNK * HP;           // [96][160]  activations pong
  float*    w4   = (float*)(bufB + MCHUNK * HP); // [96][64]   layer-4 output
  float*    dch  = w4 + MCHUNK * 64;             // [96]       distances

  int tid = threadIdx.x, lane = tid & 31, wave = tid >> 5;
  int bid = blockIdx.x;
  int b = bid / NATOMS, a = bid % NATOMS;

  const float S1 = 0.5773502692f;   // 1/sqrt(NBASIS=3)
  const float S2 = 0.0816496581f;   // 1/sqrt(H=150)

  // ---- stage weights into LDS: transposed, scaled, zero-padded ----
  for (int i = tid; i < HP * 32; i += 256) {
    int n = i >> 5, k = i & 31;
    Wt1[i] = (_Float16)((n < HDIM && k < 3) ? w1[k * HDIM + n] * S1 : 0.0f);
  }
  for (int i = tid; i < HP * HP; i += 256) {
    int n = i / HP, k = i % HP;
    bool ok = (n < HDIM && k < HDIM);
    Wt2[i] = (_Float16)(ok ? w2[k * HDIM + n] * S2 : 0.0f);
    Wt3[i] = (_Float16)(ok ? w3[k * HDIM + n] * S2 : 0.0f);
  }
  for (int i = tid; i < 64 * HP; i += 256) {
    int n = i / HP, k = i % HP;
    Wto[i] = (_Float16)((n < W && k < HDIM) ? wo[k * W + n] * S2 : 0.0f);
  }

  float ax = xyz[(b * NATOMS + a) * 3 + 0];
  float ay = xyz[(b * NATOMS + a) * 3 + 1];
  float az = xyz[(b * NATOMS + a) * 3 + 2];

  float accT = 0.0f;          // per-thread einsum partial (thread t < W)
  __syncthreads();

  for (int mc = 0; mc < 3; ++mc) {           // 3 x 96 = 288 >= 286 neighbors
    int j0 = mc * MCHUNK;
    if (tid < MCHUNK) {
      int jg = j0 + tid;
      float d = 1e30f;
      if (jg < NATOMS) {
        float dx = ax - xyz[(b * NATOMS + jg) * 3 + 0];
        float dy = ay - xyz[(b * NATOMS + jg) * 3 + 1];
        float dz = az - xyz[(b * NATOMS + jg) * 3 + 2];
        d = sqrtf(dx * dx + dy * dy + dz * dz + 1e-12f);
      }
      dch[tid] = d;
    }
    __syncthreads();
    // squared-cosine basis -> bufA cols [0,32) (K padded 3 -> 32)
    for (int i = tid; i < MCHUNK * 32; i += 256) {
      int j = i >> 5, k = i & 31;
      float v = 0.0f;
      if (k < 3) {
        float x = (dch[j] - 1.5f * k) * (1.0f / 1.5f);
        if (fabsf(x) < 1.0f) { float c = __cosf(1.5707963268f * x); v = c * c; }
      }
      bufA[j * HP + k] = (_Float16)v;
    }
    __syncthreads();
    gemm_tiles<1, true,  true >(bufA, HP, Wt1, 32, 10,     bufB, nullptr, HP, wave, lane);
    __syncthreads();
    gemm_tiles<5, true,  true >(bufB, HP, Wt2, HP, 10,     bufA, nullptr, HP, wave, lane);
    __syncthreads();
    gemm_tiles<5, true,  true >(bufA, HP, Wt3, HP, 10,     bufB, nullptr, HP, wave, lane);
    __syncthreads();
    gemm_tiles<5, false, false>(bufB, HP, Wto, HP, W >> 4, nullptr, w4, 64, wave, lane);
    __syncthreads();
    // fused masked contraction: acc[t] += sum_j mask_j * w4[j][t] * f_in[b,j,t%din]
    if (tid < W) {
      int k = tid % din;
      for (int j = 0; j < MCHUNK; ++j) {
        int jg = j0 + j;
        if (jg < NATOMS && dch[j] <= MAXR) {
          accT += w4[j * 64 + tid] * f_in[(b * NATOMS + jg) * din + k];
        }
      }
    }
    __syncthreads();
  }

  if (tid < W) w4[tid] = accT;   // reuse w4 as scratch
  __syncthreads();
  if (tid < 8) {                 // dout == 8 for all three convs
    float s = 0.0f;
    for (int k = 0; k < din; ++k) s += w4[tid * din + k];
    f_out[(b * NATOMS + a) * 8 + tid] = s * rsqrtf((float)din);
  }
}

// pool (L2 over atoms) -> linear -> batchnorm over batch -> LeakyReLU(0.2)
__global__ void head_kernel(const float* __restrict__ f1, const float* __restrict__ f2,
                            const float* __restrict__ f3,
                            const float* __restrict__ lin_w, const float* __restrict__ lin_b,
                            const float* __restrict__ bn_g, const float* __restrict__ bn_b,
                            float* __restrict__ out) {
  __shared__ float pooled[NBATCH][24];
  int t = threadIdx.x;
  if (t < 24) {
    const float* src = (t < 8) ? f1 : ((t < 16) ? f2 : f3);
    int c = t & 7;
    for (int b = 0; b < NBATCH; ++b) {
      float s = 1e-12f;
      for (int a = 0; a < NATOMS; ++a) {
        float v = src[(b * NATOMS + a) * 8 + c];
        s += v * v;
      }
      pooled[b][t] = sqrtf(s);
    }
  }
  __syncthreads();
  if (t < 24) {
    float y[NBATCH];
    for (int b = 0; b < NBATCH; ++b) {
      float s = lin_b[t];
      for (int c = 0; c < 24; ++c) s += pooled[b][c] * lin_w[t * 24 + c];
      y[b] = s;
    }
    float m   = 0.5f * (y[0] + y[1]);
    float var = 0.5f * ((y[0] - m) * (y[0] - m) + (y[1] - m) * (y[1] - m));
    float inv = rsqrtf(var + 1e-5f);
    for (int b = 0; b < NBATCH; ++b) {
      float yn = (y[b] - m) * inv * bn_g[t] + bn_b[t];
      out[b * 24 + t] = (yn > 0.0f) ? yn : 0.2f * yn;
    }
  }
}

extern "C" void kernel_launch(void* const* d_in, const int* in_sizes, int n_in,
                              void* d_out, int out_size, void* d_ws, size_t ws_size,
                              hipStream_t stream) {
  const float* xyz   = (const float*)d_in[0];
  const int*   Z     = (const int*)d_in[1];
  const float* emb   = (const float*)d_in[2];
  const float* c0w1  = (const float*)d_in[3];
  const float* c0w2  = (const float*)d_in[4];
  const float* c0w3  = (const float*)d_in[5];
  const float* c0wo  = (const float*)d_in[6];
  const float* c1w1  = (const float*)d_in[7];
  const float* c1w2  = (const float*)d_in[8];
  const float* c1w3  = (const float*)d_in[9];
  const float* c1wo  = (const float*)d_in[10];
  const float* c2w1  = (const float*)d_in[11];
  const float* c2w2  = (const float*)d_in[12];
  const float* c2w3  = (const float*)d_in[13];
  const float* c2wo  = (const float*)d_in[14];
  const float* lin_w = (const float*)d_in[15];
  const float* lin_b = (const float*)d_in[16];
  const float* bn_g  = (const float*)d_in[17];
  const float* bn_b  = (const float*)d_in[18];

  float* f0 = (float*)d_ws;                  // [B,N,4]
  float* f1 = f0 + NBATCH * NATOMS * 4;      // [B,N,8]
  float* f2 = f1 + NBATCH * NATOMS * 8;      // [B,N,8]
  float* f3 = f2 + NBATCH * NATOMS * 8;      // [B,N,8]

  int nBN = NBATCH * NATOMS;
  gather_f0_kernel<<<(nBN + 255) / 256, 256, 0, stream>>>(emb, Z, f0, nBN);

  size_t shmem = (size_t)(2 * HP * HP + 64 * HP + HP * 32 + 2 * MCHUNK * HP) * sizeof(_Float16)
               + (size_t)(MCHUNK * 64 + MCHUNK) * sizeof(float);  // ~219.5 KB < 320 KB

  conv_kernel<<<nBN, 256, shmem, stream>>>(xyz, f0, c0w1, c0w2, c0w3, c0wo, f1, 4, 32);
  conv_kernel<<<nBN, 256, shmem, stream>>>(xyz, f1, c1w1, c1w2, c1w3, c1wo, f2, 8, 64);
  conv_kernel<<<nBN, 256, shmem, stream>>>(xyz, f2, c2w1, c2w2, c2w3, c2wo, f3, 8, 64);

  head_kernel<<<1, 32, 0, stream>>>(f1, f2, f3, lin_w, lin_b, bn_g, bn_b, (float*)d_out);
}
